// FeatureSelfAttnLSTMForecast_30185030156477
// MI455X (gfx1250) — matrix operand order repaired and checked by reading, not compiled, not run on hardware
//
#include <hip/hip_runtime.h>
#include <hip/hip_bf16.h>
#include <math.h>

// Problem constants (from reference): B=256, S=512, F=64, H=256, NF=5
#define Bx  256
#define Sx  512
#define Fx  64
#define Hx  256
#define NFx 5
#define G4H 1024

typedef unsigned short u16;
typedef unsigned int  uv4  __attribute__((ext_vector_type(4)));
typedef __bf16        v16bf __attribute__((ext_vector_type(16)));
typedef float         v8f  __attribute__((ext_vector_type(8)));

union FragBF { v16bf bf; uv4 q[2]; };

__device__ __forceinline__ u16 f2bf(float f) {
  unsigned u = __float_as_uint(f);
  u += 0x7fffu + ((u >> 16) & 1u);       // round-to-nearest-even
  return (u16)(u >> 16);
}
__device__ __forceinline__ float sigm(float x) { return 1.0f / (1.0f + __expf(-x)); }

// ---------------------------------------------------------------------------
// Pack W^T (optionally [Wx^T ; Wh^T] concat along K, with K-padding for the
// x-part) into the wave32 WMMA B-fragment layout:
//   dst[((n16*NKT + kt)*32 + lane)*16 + e] = W[n][k],
//   n = n16*16 + (lane&15),  k = kt*32 + (e>>3)*16 + (lane>>4)*8 + (e&7)
// ---------------------------------------------------------------------------
__global__ void pack_weights(const float* __restrict__ Wx, int KxValid, int Kx,
                             const float* __restrict__ Wh, int Kh,
                             u16* __restrict__ dst, int N) {
  int K   = Kx + Kh;
  int NKT = K >> 5;
  long total = (long)N * K;
  for (long i = blockIdx.x * (long)blockDim.x + threadIdx.x; i < total;
       i += (long)gridDim.x * blockDim.x) {
    int  e    = (int)(i & 15);
    int  lane = (int)((i >> 4) & 31);
    long rest = i >> 9;
    int  kt   = (int)(rest % NKT);
    int  n16  = (int)(rest / NKT);
    int  n    = n16 * 16 + (lane & 15);
    int  kl   = kt * 32 + ((e >> 3) << 4) + ((lane >> 4) << 3) + (e & 7);
    float v;
    if (kl < Kx) v = (kl < KxValid) ? Wx[(long)n * KxValid + kl] : 0.0f;
    else         v = Wh[(long)n * Kh + (kl - Kx)];
    dst[i] = f2bf(v);
  }
}

__global__ void combine_bias(const float* a0, const float* c0, float* d0,
                             const float* a1, const float* c1, float* d1,
                             const float* a2, const float* c2, float* d2,
                             const float* a3, const float* c3, float* d3) {
  int i = blockIdx.x * blockDim.x + threadIdx.x;
  if (i < G4H) {
    d0[i] = a0[i] + c0[i];
    d1[i] = a1[i] + c1[i];
    d2[i] = a2[i] + c2[i];
    d3[i] = a3[i] + c3[i];
  }
}

// ---------------------------------------------------------------------------
// Feature self-attention:  e = [h1 | x_t] @ W_ia^T + b_ia ;  a = softmax(e);
// xatt = a * x_t   (written bf16).  One block per 16-row M tile; 4 waves
// cover the 64 output features.  WMMA 16x16x32 bf16, K = 320.
// ---------------------------------------------------------------------------
__global__ void __launch_bounds__(128)
attn_kernel(const float* __restrict__ x, int t,
            const u16* __restrict__ H1,       // B x H bf16 (h1_prev)
            const u16* __restrict__ BiaP,     // packed 320x64
            const float* __restrict__ b_ia,
            u16* __restrict__ XATT) {
  const int Ktot = Hx + Fx;   // 320
  const int AP   = Ktot + 8;  // row pad to spread LDS banks
  __shared__ __align__(16) u16 a_sh[16 * 328];
  __shared__ float e_sh[16 * 64];
  int m0  = blockIdx.x * 16;
  int tid = threadIdx.x;

  for (int idx = tid; idx < 16 * Ktot; idx += 128) {
    int r = idx / Ktot, k = idx - r * Ktot;
    int b = m0 + r;
    u16 v = (k < Hx) ? H1[b * Hx + k]
                     : f2bf(x[(long)b * Sx * Fx + (long)t * Fx + (k - Hx)]);
    a_sh[r * AP + k] = v;
  }
  __syncthreads();

  int w = tid >> 5, lane = tid & 31;
  int ml = lane & 15, hi = lane >> 4;
  const int NKT = Ktot / 32;  // 10
  v8f acc = {};
  #pragma unroll
  for (int kt = 0; kt < NKT; ++kt) {
    FragBF a, b;
    a.q[0] = *(const uv4*)&a_sh[ml * AP + kt * 32 + hi * 8];
    a.q[1] = *(const uv4*)&a_sh[ml * AP + kt * 32 + 16 + hi * 8];
    const uv4* bp = (const uv4*)(BiaP + ((long)(w * NKT + kt) * 32 + lane) * 16);
    b.q[0] = bp[0]; b.q[1] = bp[1];
    acc = __builtin_amdgcn_wmma_f32_16x16x32_bf16(false, a.bf, false, b.bf,
                                                  (short)0, acc, false, false);
  }
  int n = w * 16 + ml;
  float bias = b_ia[n];
  #pragma unroll
  for (int r = 0; r < 8; ++r) {
    int m = r + hi * 8;
    e_sh[m * 64 + n] = acc[r] + bias;
  }
  __syncthreads();

  if (tid < 16) {  // row-wise softmax over 64 features
    float mx = -1e30f;
    for (int f = 0; f < 64; ++f) mx = fmaxf(mx, e_sh[tid * 64 + f]);
    float s = 0.f;
    for (int f = 0; f < 64; ++f) {
      float ev = __expf(e_sh[tid * 64 + f] - mx);
      e_sh[tid * 64 + f] = ev; s += ev;
    }
    float inv = 1.0f / s;
    for (int f = 0; f < 64; ++f) e_sh[tid * 64 + f] *= inv;
  }
  __syncthreads();

  for (int idx = tid; idx < 16 * 64; idx += 128) {
    int r = idx >> 6, f = idx & 63;
    int b = m0 + r;
    float xa = e_sh[idx] * x[(long)b * Sx * Fx + (long)t * Fx + f];
    XATT[b * 64 + f] = f2bf(xa);
  }
}

// ---------------------------------------------------------------------------
// Fused LSTM step: gates = [srcX | srcH] @ Wpacked + bias; cell update.
// Grid (16 m-tiles, 16 j-tiles of H); 4 waves/block = gates i,f,g,o of the
// same 16x16 h-slice; waves 1..3 push their gate tiles through LDS, wave 0
// performs the cell nonlinearity and writes c (fp32) and h (bf16 [+fp32]).
// ---------------------------------------------------------------------------
template <int KX, int KH>
__global__ void __launch_bounds__(128)
lstm_kernel(const u16* __restrict__ srcX,
            const u16* __restrict__ srcH,
            const u16* __restrict__ Bp,       // packed (KX+KH) x 1024
            const float* __restrict__ bias4h, // 4H fp32 (b_ih+b_hh)
            float* __restrict__ C,            // B x H fp32 (in-place)
            u16* __restrict__ Hout,           // B x H bf16
            float* __restrict__ Hf32) {       // optional B x H fp32
  constexpr int K   = KX + KH;
  constexpr int AP  = K + 8;
  constexpr int NKT = K >> 5;
  __shared__ __align__(16) u16 a_sh[16 * (512 + 8)];
  __shared__ float xch[3 * 256];

  int m0  = blockIdx.x * 16;
  int j16 = blockIdx.y;
  int tid = threadIdx.x;

  for (int idx = tid; idx < 16 * K; idx += 128) {
    int r = idx / K, k = idx - r * K;
    int b = m0 + r;
    a_sh[r * AP + k] = (k < KX) ? srcX[b * KX + k] : srcH[b * KH + (k - KX)];
  }
  __syncthreads();

  int w = tid >> 5, lane = tid & 31;
  int ml = lane & 15, hi = lane >> 4;
  int n16g = w * (Hx / 16) + j16;            // global 16-col tile in [0,64)
  const u16* bbase = Bp + (long)n16g * NKT * 32 * 16;

  v8f acc = {};
  #pragma unroll
  for (int kt = 0; kt < NKT; ++kt) {
    FragBF a, b;
    a.q[0] = *(const uv4*)&a_sh[ml * AP + kt * 32 + hi * 8];
    a.q[1] = *(const uv4*)&a_sh[ml * AP + kt * 32 + 16 + hi * 8];
    const uv4* bp = (const uv4*)(bbase + ((long)kt * 32 + lane) * 16);
    __builtin_prefetch((const void*)(bp + 64), 0, 1);   // global_prefetch_b8
    b.q[0] = bp[0]; b.q[1] = bp[1];
    acc = __builtin_amdgcn_wmma_f32_16x16x32_bf16(false, a.bf, false, b.bf,
                                                  (short)0, acc, false, false);
  }

  float bias = bias4h[w * Hx + j16 * 16 + ml];
  if (w != 0) {
    #pragma unroll
    for (int r = 0; r < 8; ++r) {
      int m = r + hi * 8;
      xch[(w - 1) * 256 + m * 16 + ml] = acc[r] + bias;
    }
  }
  __syncthreads();
  if (w == 0) {
    #pragma unroll
    for (int r = 0; r < 8; ++r) {
      int m  = r + hi * 8;
      int b  = m0 + m;
      int j  = j16 * 16 + ml;
      float gi = acc[r] + bias;
      float gf = xch[0 * 256 + m * 16 + ml];
      float gg = xch[1 * 256 + m * 16 + ml];
      float go = xch[2 * 256 + m * 16 + ml];
      float c_old = C[b * Hx + j];
      float c_new = sigm(gf) * c_old + sigm(gi) * tanhf(gg);
      float h_new = sigm(go) * tanhf(c_new);
      C[b * Hx + j]    = c_new;
      Hout[b * Hx + j] = f2bf(h_new);
      if (Hf32) Hf32[b * Hx + j] = h_new;
    }
  }
}

__global__ void dec_setup(const float* __restrict__ x, u16* __restrict__ INP) {
  int i = blockIdx.x * blockDim.x + threadIdx.x;
  if (i < Bx * 32) {
    int b = i >> 5, c = i & 31;
    INP[i] = (c == 0) ? f2bf(x[(long)b * Sx * Fx + (long)(Sx - 1) * Fx + 0])
                      : (u16)0;
  }
}

__global__ void fc_kernel(const float* __restrict__ H1f,
                          const float* __restrict__ Wfc,
                          const float* __restrict__ bfc,
                          float* __restrict__ out, int step,
                          u16* __restrict__ INP) {
  int b = threadIdx.x;  // 256 threads, one batch row each
  float s = bfc[0];
  for (int k = 0; k < Hx; ++k) s += H1f[b * Hx + k] * Wfc[k];
  out[b * NFx + step] = s;
  INP[b * 32] = f2bf(s);   // feed next decoder step (cols 1..31 stay zero)
}

// ---------------------------------------------------------------------------
extern "C" void kernel_launch(void* const* d_in, const int* in_sizes, int n_in,
                              void* d_out, int out_size, void* d_ws, size_t ws_size,
                              hipStream_t stream) {
  const float* x      = (const float*)d_in[0];
  const float* W_ia   = (const float*)d_in[1];
  const float* b_ia   = (const float*)d_in[2];
  const float* W_ih0  = (const float*)d_in[3];
  const float* W_hh0  = (const float*)d_in[4];
  const float* b_ih0  = (const float*)d_in[5];
  const float* b_hh0  = (const float*)d_in[6];
  const float* W_ih1  = (const float*)d_in[7];
  const float* W_hh1  = (const float*)d_in[8];
  const float* b_ih1  = (const float*)d_in[9];
  const float* b_hh1  = (const float*)d_in[10];
  // d_in[11..13] = W_attn, b_attn, v_w: the context they produce is dead code
  // in the reference (never feeds the output) -> eliminated.
  const float* dW_ih0 = (const float*)d_in[14];
  const float* dW_hh0 = (const float*)d_in[15];
  const float* db_ih0 = (const float*)d_in[16];
  const float* db_hh0 = (const float*)d_in[17];
  const float* dW_ih1 = (const float*)d_in[18];
  const float* dW_hh1 = (const float*)d_in[19];
  const float* db_ih1 = (const float*)d_in[20];
  const float* db_hh1 = (const float*)d_in[21];
  const float* W_fc   = (const float*)d_in[22];
  const float* b_fc   = (const float*)d_in[23];
  float* out = (float*)d_out;

  char* ws = (char*)d_ws;
  size_t off = 0;
  auto alloc = [&](size_t bytes) {
    char* p = ws + off;
    off += (bytes + 255) & ~(size_t)255;
    return p;
  };
  u16*   BiaP = (u16*)alloc((size_t)320 * 64 * 2);
  u16*   B0P  = (u16*)alloc((size_t)320 * G4H * 2);
  u16*   B1P  = (u16*)alloc((size_t)512 * G4H * 2);
  u16*   Bd0P = (u16*)alloc((size_t)288 * G4H * 2);
  u16*   Bd1P = (u16*)alloc((size_t)512 * G4H * 2);
  float* bc0  = (float*)alloc(G4H * 4);
  float* bc1  = (float*)alloc(G4H * 4);
  float* bd0  = (float*)alloc(G4H * 4);
  float* bd1  = (float*)alloc(G4H * 4);
  float* C0   = (float*)alloc((size_t)Bx * Hx * 4);
  float* C1   = (float*)alloc((size_t)Bx * Hx * 4);
  u16*   H0a  = (u16*)alloc((size_t)Bx * Hx * 2);
  u16*   H0b  = (u16*)alloc((size_t)Bx * Hx * 2);
  u16*   H1a  = (u16*)alloc((size_t)Bx * Hx * 2);
  u16*   H1b  = (u16*)alloc((size_t)Bx * Hx * 2);
  u16*   XATT = (u16*)alloc((size_t)Bx * 64 * 2);
  u16*   INP  = (u16*)alloc((size_t)Bx * 32 * 2);
  float* H1F  = (float*)alloc((size_t)Bx * Hx * 4);

  // ---- one-time prep (cheap; re-run every call for determinism) ----
  pack_weights<<<256, 256, 0, stream>>>(W_ia,   320, 320, nullptr,   0, BiaP,  64);
  pack_weights<<<256, 256, 0, stream>>>(W_ih0,   64,  64, W_hh0,   256, B0P,  G4H);
  pack_weights<<<256, 256, 0, stream>>>(W_ih1,  256, 256, W_hh1,   256, B1P,  G4H);
  pack_weights<<<256, 256, 0, stream>>>(dW_ih0,   1,  32, dW_hh0,  256, Bd0P, G4H);
  pack_weights<<<256, 256, 0, stream>>>(dW_ih1, 256, 256, dW_hh1,  256, Bd1P, G4H);
  combine_bias<<<4, 256, 0, stream>>>(b_ih0, b_hh0, bc0, b_ih1, b_hh1, bc1,
                                      db_ih0, db_hh0, bd0, db_ih1, db_hh1, bd1);
  hipMemsetAsync(C0, 0, (size_t)Bx * Hx * 4, stream);
  hipMemsetAsync(C1, 0, (size_t)Bx * Hx * 4, stream);
  hipMemsetAsync(H0a, 0, (size_t)Bx * Hx * 2, stream);
  hipMemsetAsync(H0b, 0, (size_t)Bx * Hx * 2, stream);
  hipMemsetAsync(H1a, 0, (size_t)Bx * Hx * 2, stream);
  hipMemsetAsync(H1b, 0, (size_t)Bx * Hx * 2, stream);

  dim3 lgrid(Bx / 16, Hx / 16);  // 16 x 16 blocks

  // ---- encoder: 512 sequential steps ----
  for (int t = 0; t < Sx; ++t) {
    int p = t & 1;
    u16* H0r = p ? H0b : H0a; u16* H0w = p ? H0a : H0b;
    u16* H1r = p ? H1b : H1a; u16* H1w = p ? H1a : H1b;
    attn_kernel<<<Bx / 16, 128, 0, stream>>>(x, t, H1r, BiaP, b_ia, XATT);
    lstm_kernel<64, 256><<<lgrid, 128, 0, stream>>>(XATT, H0r, B0P, bc0, C0, H0w, nullptr);
    lstm_kernel<256, 256><<<lgrid, 128, 0, stream>>>(H0w, H1r, B1P, bc1, C1, H1w, nullptr);
  }
  // encoder finals now live in H0a / H1a (t=511 wrote parity-0 buffers), C0, C1

  // ---- decoder: 5 steps ----
  dec_setup<<<(Bx * 32) / 256, 256, 0, stream>>>(x, INP);
  for (int d = 0; d < NFx; ++d) {
    int p = d & 1;
    u16* H0r = p ? H0b : H0a; u16* H0w = p ? H0a : H0b;
    u16* H1r = p ? H1b : H1a; u16* H1w = p ? H1a : H1b;
    lstm_kernel<32, 256><<<lgrid, 128, 0, stream>>>(INP, H0r, Bd0P, bd0, C0, H0w, nullptr);
    lstm_kernel<256, 256><<<lgrid, 128, 0, stream>>>(H0w, H1r, Bd1P, bd1, C1, H1w, H1F);
    fc_kernel<<<1, 256, 0, stream>>>(H1F, W_fc, b_fc, out, d, INP);
  }
}
